// WaveNet_3410204033500
// MI455X (gfx1250) — matrix operand order repaired and checked by reading, not compiled, not run on hardware
//
#include <hip/hip_runtime.h>

// ---------------------------------------------------------------------------
// WaveNet vocoder forward for MI455X (gfx1250, wave32, WMMA 16x16x32 f16)
// 32 rows per workgroup, fully unrolled WMMA pipelines.
// ---------------------------------------------------------------------------

typedef __attribute__((ext_vector_type(16))) _Float16 v16h;
typedef __attribute__((ext_vector_type(8)))  _Float16 v8h;
typedef __attribute__((ext_vector_type(8)))  float    v8f;

#define MTOT   131072L   // B * T_audio = 8 * 16384
#define TAUD   16384
#define NBLK   20

__device__ __forceinline__ v8f zero_v8f() {
  v8f r;
#pragma unroll
  for (int i = 0; i < 8; ++i) r[i] = 0.0f;
  return r;
}

__device__ __forceinline__ v16h zero_v16h() {
  v16h r;
#pragma unroll
  for (int i = 0; i < 16; ++i) r[i] = (_Float16)0.0f;
  return r;
}

// A-fragment (16x32 f16) from row-major f16, row = mbase + (lane&15).
// lane<16 holds K = {0..7, 16..23}+kbase ; lane>=16 holds K = {8..15, 24..31}.
__device__ __forceinline__ v16h make_a(const _Float16* p) {
  v8h lo = *(const v8h*)p;          // elements 0..7
  v8h hi = *(const v8h*)(p + 16);   // elements 8..15
  v16h r;
#pragma unroll
  for (int i = 0; i < 8; ++i) { r[i] = lo[i]; r[i + 8] = hi[i]; }
  return r;
}

__device__ __forceinline__ v8f wmma_f16(v16h a, v16h b, v8f c) {
  return __builtin_amdgcn_wmma_f32_16x16x32_f16(false, a, false, b,
                                                (short)0, c, false, false);
}

// ---------------------------------------------------------------------------
// Pack a row-major f32 weight [Ksrc, N] into WMMA B-fragments (f16).
// frag = kt*(N/16) + nt ; element i = lane*16 + e with
// K = kt*32 + (e>=8 ? 16 : 0) + (lane>=16 ? 8 : 0) + (e&7), Ncol = nt*16 + (lane&15).
// gridDim.x = (Kpad/32) * (N/16).
// ---------------------------------------------------------------------------
__global__ __launch_bounds__(256)
void pack_frags(const float* __restrict__ src, _Float16* __restrict__ dst,
                int Ksrc, int N) {
  const int ntiles = N >> 4;
  const int frag   = blockIdx.x;
  const int kt     = frag / ntiles;
  const int nt     = frag - kt * ntiles;
  for (int i = threadIdx.x; i < 512; i += 256) {
    const int lane = i >> 4;
    const int e    = i & 15;
    const int k    = kt * 32 + ((e & 8) ? 16 : 0) + ((lane & 16) ? 8 : 0) + (e & 7);
    const int n    = nt * 16 + (lane & 15);
    const float v  = (k < Ksrc) ? src[(long)k * N + n] : 0.0f;
    dst[(long)frag * 512 + i] = (_Float16)v;
  }
}

__global__ void zero_regions(unsigned* __restrict__ a, long na,
                             unsigned* __restrict__ b, long nb) {
  const long stride = (long)gridDim.x * blockDim.x;
  const long t0 = (long)blockIdx.x * blockDim.x + threadIdx.x;
  for (long i = t0; i < na; i += stride) a[i] = 0u;
  for (long i = t0; i < nb; i += stride) b[i] = 0u;
}

__global__ void f32_to_f16(const float* __restrict__ s,
                           _Float16* __restrict__ d, long n) {
  const long stride = (long)gridDim.x * blockDim.x;
  for (long i = (long)blockIdx.x * blockDim.x + threadIdx.x; i < n; i += stride)
    d[i] = (_Float16)s[i];
}

__global__ void relu_f32_to_f16(const float* __restrict__ s,
                                _Float16* __restrict__ d, long n) {
  const long stride = (long)gridDim.x * blockDim.x;
  for (long i = (long)blockIdx.x * blockDim.x + threadIdx.x; i < n; i += stride)
    d[i] = (_Float16)fmaxf(s[i], 0.0f);
}

// ---------------------------------------------------------------------------
// Conditioning upsampler: 3 chained transposed convs (4x, 8x, 8x) per
// (batch, mel) channel, staged in LDS, output f16 [B*T, 96].
// ---------------------------------------------------------------------------
__global__ __launch_bounds__(256)
void upsample_cond(const float* __restrict__ c, const float* __restrict__ w0,
                   const float* __restrict__ w1, const float* __restrict__ w2,
                   _Float16* __restrict__ cond) {
  __shared__ __align__(16) float s0[64];
  __shared__ __align__(16) float s1[256];
  __shared__ __align__(16) float s2[2048];
  const int b   = blockIdx.x / 80;
  const int mel = blockIdx.x - b * 80;
  const int tid = threadIdx.x;

  if (tid < 64) s0[tid] = c[((long)b * 80 + mel) * 64 + tid];
  __syncthreads();

  for (int o = tid; o < 256; o += 256) {   // stride 4, K=8, pad_lo=5
    float acc = 0.0f;
#pragma unroll
    for (int k = 0; k < 8; ++k) {
      const int jd = o + k - 5;
      if (jd >= 0 && jd <= 63 * 4 && (jd & 3) == 0) acc += w0[k] * s0[jd >> 2];
    }
    s1[o] = acc;
  }
  __syncthreads();

  for (int o = tid; o < 2048; o += 256) {  // stride 8, K=16, pad_lo=11
    float acc = 0.0f;
#pragma unroll
    for (int k = 0; k < 16; ++k) {
      const int jd = o + k - 11;
      if (jd >= 0 && jd <= 255 * 8 && (jd & 7) == 0) acc += w1[k] * s1[jd >> 3];
    }
    s2[o] = acc;
  }
  __syncthreads();

  for (int o = tid; o < TAUD; o += 256) {  // stride 8, K=16 ; -> cond[*,96]
    float acc = 0.0f;
#pragma unroll
    for (int k = 0; k < 16; ++k) {
      const int jd = o + k - 11;
      if (jd >= 0 && jd <= 2047 * 8 && (jd & 7) == 0) acc += w2[k] * s2[jd >> 3];
    }
    cond[((long)b * TAUD + o) * 96 + mel] = (_Float16)acc;
  }
}

// ---------------------------------------------------------------------------
// Templated WMMA GEMM, fully unrolled. 32 rows (2 M-tiles) per workgroup,
// 8 waves; wave w owns N-tiles {w} or {w, w+8}. Every B fragment feeds two
// independent accumulator chains (one per M-tile) for WMMA latency hiding.
// ---------------------------------------------------------------------------
template <int KT, int NT, bool F32OUT, bool RELU>
__global__ __launch_bounds__(256)
void gemm16(const _Float16* __restrict__ A, int lda,
            const _Float16* __restrict__ Bfrag, const float* __restrict__ bias,
            void* __restrict__ outp, int ldo) {
  constexpr int NPW = NT / 8;  // N-tiles per wave (1 or 2)
  const int  tid    = threadIdx.x;
  const int  lane   = tid & 31;
  const int  wave   = tid >> 5;
  const long mbase  = (long)blockIdx.x * 32;
  const int  m_lane = lane & 15;
  const int  hi16   = (lane >> 4) & 1;

  v8f acc[2][NPW];
#pragma unroll
  for (int mt = 0; mt < 2; ++mt)
#pragma unroll
    for (int j = 0; j < NPW; ++j) acc[mt][j] = zero_v8f();

#pragma unroll
  for (int kt = 0; kt < KT; ++kt) {
    const v16h a0 = make_a(A + (mbase + m_lane) * lda + kt * 32 + hi16 * 8);
    const v16h a1 = make_a(A + (mbase + 16 + m_lane) * lda + kt * 32 + hi16 * 8);
#pragma unroll
    for (int j = 0; j < NPW; ++j) {
      const v16h b =
          *(const v16h*)(Bfrag + ((long)(kt * NT + wave + j * 8)) * 512 + lane * 16);
      acc[0][j] = wmma_f16(a0, b, acc[0][j]);
      acc[1][j] = wmma_f16(a1, b, acc[1][j]);
    }
  }

#pragma unroll
  for (int j = 0; j < NPW; ++j) {
    const int   col = (wave + j * 8) * 16 + m_lane;
    const float bv  = bias[col];
#pragma unroll
    for (int mt = 0; mt < 2; ++mt)
#pragma unroll
      for (int e = 0; e < 8; ++e) {
        const long row = mbase + mt * 16 + e + hi16 * 8;
        float v = acc[mt][j][e] + bv;
        if (RELU) v = fmaxf(v, 0.0f);
        if (F32OUT) ((float*)outp)[row * ldo + col] = v;
        else        ((_Float16*)outp)[row * ldo + col] = (_Float16)v;
      }
  }
}

// ---------------------------------------------------------------------------
// One residual block. Workgroup = 32 time rows (2 M-tiles), 8 waves.
//   g    = dilated causal conv (3 taps, K=128 -> G=256) + cond GEMM (K=96)
//   z    = tanh(a) * sigmoid(b)            (GLU over LDS-staged g)
//   skip += z @ Ws + bs                    (f32 RMW, L2-resident)
//   xout = (xin + z @ Wo + bo) * sqrt(.5)  (f16 ping-pong buffer)
// 76 static WMMAs per workgroup; 4 independent chains in the conv stage.
// ---------------------------------------------------------------------------
__global__ __launch_bounds__(256)
void res_block(const _Float16* __restrict__ xin, _Float16* __restrict__ xout,
               const _Float16* __restrict__ cond, float* __restrict__ skip,
               const _Float16* __restrict__ wd_frag,   // 3 taps * 64 frags
               const _Float16* __restrict__ wc_frag,   // 3 kt  * 16 frags
               const _Float16* __restrict__ ws_frag,   // 4 kt  * 8 frags
               const _Float16* __restrict__ wo_frag,   // 4 kt  * 8 frags
               const float* __restrict__ bd, const float* __restrict__ bs,
               const float* __restrict__ bo, int dil) {
  __shared__ __align__(16) float    lds_g[32][256];   // 32 KB
  __shared__ __align__(16) _Float16 lds_z[32][128];   //  8 KB

  const int  tid    = threadIdx.x;
  const int  lane   = tid & 31;
  const int  wave   = tid >> 5;
  const long mbase  = (long)blockIdx.x * 32;
  const int  m_lane = lane & 15;
  const int  hi16   = (lane >> 4) & 1;
  const int  t0     = (int)(mbase & (TAUD - 1));  // 32 | 16384, same batch

  // ---- stage 1: gated dilated conv + conditioning ----
  v8f acc[2][2];  // [mtile][ntile j]  (ntiles wave, wave+8)
#pragma unroll
  for (int mt = 0; mt < 2; ++mt)
#pragma unroll
    for (int j = 0; j < 2; ++j) acc[mt][j] = zero_v8f();

#pragma unroll
  for (int tap = 0; tap < 3; ++tap) {
#pragma unroll
    for (int kt = 0; kt < 4; ++kt) {
      v16h a[2];
#pragma unroll
      for (int mt = 0; mt < 2; ++mt) {
        const int  t_in  = t0 + mt * 16 + m_lane;
        const bool valid = (t_in - tap * dil) >= 0;  // causal zero-pad
        if (valid) {
          const long row = mbase + mt * 16 + m_lane - (long)tap * dil;
          a[mt] = make_a(xin + row * 128 + kt * 32 + hi16 * 8);
        } else {
          a[mt] = zero_v16h();
        }
      }
      const _Float16* bp = wd_frag + ((long)(tap * 4 + kt) * 16) * 512;
      const v16h b0 = *(const v16h*)(bp + (long)wave * 512 + lane * 16);
      const v16h b1 = *(const v16h*)(bp + (long)(wave + 8) * 512 + lane * 16);
      acc[0][0] = wmma_f16(a[0], b0, acc[0][0]);
      acc[1][0] = wmma_f16(a[1], b0, acc[1][0]);
      acc[0][1] = wmma_f16(a[0], b1, acc[0][1]);
      acc[1][1] = wmma_f16(a[1], b1, acc[1][1]);
    }
  }
#pragma unroll
  for (int kt = 0; kt < 3; ++kt) {
    v16h a[2];
#pragma unroll
    for (int mt = 0; mt < 2; ++mt)
      a[mt] = make_a(cond + (mbase + mt * 16 + m_lane) * 96 + kt * 32 + hi16 * 8);
    const _Float16* bp = wc_frag + ((long)kt * 16) * 512;
    const v16h b0 = *(const v16h*)(bp + (long)wave * 512 + lane * 16);
    const v16h b1 = *(const v16h*)(bp + (long)(wave + 8) * 512 + lane * 16);
    acc[0][0] = wmma_f16(a[0], b0, acc[0][0]);
    acc[1][0] = wmma_f16(a[1], b0, acc[1][0]);
    acc[0][1] = wmma_f16(a[0], b1, acc[0][1]);
    acc[1][1] = wmma_f16(a[1], b1, acc[1][1]);
  }
#pragma unroll
  for (int mt = 0; mt < 2; ++mt)
#pragma unroll
    for (int e = 0; e < 8; ++e) {
      const int m = mt * 16 + e + hi16 * 8;
      lds_g[m][wave * 16 + m_lane]       = acc[mt][0][e];
      lds_g[m][(wave + 8) * 16 + m_lane] = acc[mt][1][e];
    }
  __syncthreads();

  // ---- stage 2: GLU  z = tanh(a + bd_a) * sigmoid(b + bd_b) ----
  {
    const int col = tid & 127;
    const int r0  = (tid >> 7) * 16;
    const float ba = bd[col];
    const float bb = bd[col + 128];
#pragma unroll
    for (int r = 0; r < 16; ++r) {
      const float av = lds_g[r0 + r][col] + ba;
      const float gv = lds_g[r0 + r][col + 128] + bb;
      const float z  = tanhf(av) / (1.0f + __expf(-gv));
      lds_z[r0 + r][col] = (_Float16)z;
    }
  }
  __syncthreads();

  // ---- stage 3: skip and residual GEMMs (K=128, N=128; wave -> one N-tile)
  const int   col = wave * 16 + m_lane;
  const float vbs = bs[col];
  const float vbo = bo[col];
#pragma unroll
  for (int mt = 0; mt < 2; ++mt) {
    v8f accS = zero_v8f();
    v8f accR = zero_v8f();
#pragma unroll
    for (int kt = 0; kt < 4; ++kt) {
      const v16h a  = make_a(&lds_z[mt * 16 + m_lane][kt * 32 + hi16 * 8]);
      const v16h bS = *(const v16h*)(ws_frag + ((long)(kt * 8 + wave)) * 512 + lane * 16);
      const v16h bR = *(const v16h*)(wo_frag + ((long)(kt * 8 + wave)) * 512 + lane * 16);
      accS = wmma_f16(a, bS, accS);
      accR = wmma_f16(a, bR, accR);
    }
#pragma unroll
    for (int e = 0; e < 8; ++e) {
      const long row = mbase + mt * 16 + e + hi16 * 8;
      const long idx = row * 128 + col;
      skip[idx] = skip[idx] + accS[e] + vbs;
      const float xo = ((float)xin[idx] + accR[e] + vbo) * 0.70710678118654752f;
      xout[idx] = (_Float16)xo;
    }
  }
}

// ---------------------------------------------------------------------------
// Host-side orchestration
// ---------------------------------------------------------------------------
extern "C" void kernel_launch(void* const* d_in, const int* in_sizes, int n_in,
                              void* d_out, int out_size, void* d_ws, size_t ws_size,
                              hipStream_t stream) {
  (void)in_sizes; (void)n_in; (void)out_size; (void)ws_size;

  const float* inputs  = (const float*)d_in[0];
  const float* c       = (const float*)d_in[1];
  const float* up_w0   = (const float*)d_in[2];
  const float* up_w1   = (const float*)d_in[3];
  const float* up_w2   = (const float*)d_in[4];
  const float* first_w = (const float*)d_in[5];
  const float* first_b = (const float*)d_in[6];
  const float* Wd      = (const float*)d_in[7];
  const float* bd      = (const float*)d_in[8];
  const float* Wc      = (const float*)d_in[9];
  const float* Ws      = (const float*)d_in[10];
  const float* bs      = (const float*)d_in[11];
  const float* Wo      = (const float*)d_in[12];
  const float* bo      = (const float*)d_in[13];
  const float* fin1_w  = (const float*)d_in[14];
  const float* fin1_b  = (const float*)d_in[15];
  const float* fin2_w  = (const float*)d_in[16];
  const float* fin2_b  = (const float*)d_in[17];

  // ---- workspace layout (bytes) ----
  char* ws = (char*)d_ws;
  const long OFF_FIRST = 0;                          //   64 frags
  const long OFF_WD    = OFF_FIRST + 64L * 1024;     // 3840 frags
  const long OFF_WC    = OFF_WD    + 3840L * 1024;   //  960 frags
  const long OFF_WS    = OFF_WC    + 960L * 1024;    //  640 frags
  const long OFF_WO    = OFF_WS    + 640L * 1024;    //  640 frags
  const long OFF_F1    = OFF_WO    + 640L * 1024;    //   32 frags
  const long OFF_F2    = OFF_F1    + 32L * 1024;     //   64 frags
  const long OFF_INF16 = OFF_F2    + 64L * 1024;     // 67,108,864 (reused)
  const long OFF_XA    = OFF_INF16 + MTOT * 256 * 2; // 33,554,432
  const long OFF_XB    = OFF_XA    + MTOT * 128 * 2; // 33,554,432
  const long OFF_COND  = OFF_XB    + MTOT * 128 * 2; // 25,165,824
  const long OFF_SKIP  = OFF_COND  + MTOT * 96 * 2;  // 67,108,864
  const long OFF_RS    = OFF_INF16;                  // relu(skip) f16
  const long OFF_H     = OFF_INF16 + MTOT * 128 * 2; // hidden f16

  _Float16* wf_first = (_Float16*)(ws + OFF_FIRST);
  _Float16* wf_wd    = (_Float16*)(ws + OFF_WD);
  _Float16* wf_wc    = (_Float16*)(ws + OFF_WC);
  _Float16* wf_ws    = (_Float16*)(ws + OFF_WS);
  _Float16* wf_wo    = (_Float16*)(ws + OFF_WO);
  _Float16* wf_f1    = (_Float16*)(ws + OFF_F1);
  _Float16* wf_f2    = (_Float16*)(ws + OFF_F2);
  _Float16* in_f16   = (_Float16*)(ws + OFF_INF16);
  _Float16* xA       = (_Float16*)(ws + OFF_XA);
  _Float16* xB       = (_Float16*)(ws + OFF_XB);
  _Float16* condb    = (_Float16*)(ws + OFF_COND);
  float*    skip     = (float*)   (ws + OFF_SKIP);
  _Float16* rs       = (_Float16*)(ws + OFF_RS);
  _Float16* hbuf     = (_Float16*)(ws + OFF_H);

  const dim3 blk(256);

  // ---- 1) pack weights into WMMA B-fragment layout ----
  pack_frags<<<8 * 8, blk, 0, stream>>>(first_w, wf_first, 256, 128);
  pack_frags<<<(20 * 3 * 128 / 32) * 16, blk, 0, stream>>>(Wd, wf_wd, 20 * 3 * 128, 256);
  for (int i = 0; i < NBLK; ++i)  // K padded 80 -> 96 with zeros
    pack_frags<<<3 * 16, blk, 0, stream>>>(Wc + (long)i * 80 * 256,
                                           wf_wc + (long)i * 48 * 512, 80, 256);
  pack_frags<<<(20 * 128 / 32) * 8, blk, 0, stream>>>(Ws, wf_ws, 20 * 128, 128);
  pack_frags<<<(20 * 128 / 32) * 8, blk, 0, stream>>>(Wo, wf_wo, 20 * 128, 128);
  pack_frags<<<4 * 8, blk, 0, stream>>>(fin1_w, wf_f1, 128, 128);
  pack_frags<<<4 * 16, blk, 0, stream>>>(fin2_w, wf_f2, 128, 256);

  // ---- 2) zero skip + cond ; upsample cond ; convert inputs ----
  zero_regions<<<8192, blk, 0, stream>>>((unsigned*)skip, MTOT * 128,
                                         (unsigned*)condb, MTOT * 96 / 2);
  upsample_cond<<<8 * 80, blk, 0, stream>>>(c, up_w0, up_w1, up_w2, condb);
  f32_to_f16<<<8192, blk, 0, stream>>>(inputs, in_f16, MTOT * 256);

  // ---- 3) first 1x1 conv: [M,256] @ [256,128] -> xA ----
  gemm16<8, 8, false, false><<<MTOT / 32, blk, 0, stream>>>(
      in_f16, 256, wf_first, first_b, xA, 128);

  // ---- 4) 20 residual blocks (ping-pong xA/xB) ----
  _Float16* xi = xA;
  _Float16* xo = xB;
  for (int i = 0; i < NBLK; ++i) {
    const int d = 1 << (i % 10);
    res_block<<<MTOT / 32, blk, 0, stream>>>(
        xi, xo, condb, skip,
        wf_wd + (long)i * 192 * 512, wf_wc + (long)i * 48 * 512,
        wf_ws + (long)i * 32 * 512,  wf_wo + (long)i * 32 * 512,
        bd + (long)i * 256, bs + (long)i * 128, bo + (long)i * 128, d);
    _Float16* t = xi; xi = xo; xo = t;
  }

  // ---- 5) final stack: relu -> 1x1(128) -> relu -> 1x1(256) ----
  relu_f32_to_f16<<<8192, blk, 0, stream>>>(skip, rs, MTOT * 128);
  gemm16<4, 8, false, true><<<MTOT / 32, blk, 0, stream>>>(
      rs, 128, wf_f1, fin1_b, hbuf, 128);
  gemm16<4, 16, true, false><<<MTOT / 32, blk, 0, stream>>>(
      hbuf, 128, wf_f2, fin2_b, (float*)d_out, 256);
}
